// VerificationLayer_49984829391047
// MI455X (gfx1250) — 8-bit matrix operand order repaired and checked by reading, compiled, not run on hardware
//
#include <hip/hip_runtime.h>


typedef __attribute__((ext_vector_type(8))) int v8i;

#define NKI   4096
#define NTILE 2048

__device__ __forceinline__ unsigned spread4(unsigned x) {
    return (x * 0x00204081u) & 0x01010101u;
}
__device__ __forceinline__ unsigned xtime(unsigned p) {
    return ((p << 1) & 255u) ^ ((p >> 7) * 0x1Du);
}

__global__ void __launch_bounds__(256) gf_zero(unsigned* st, int n) {
    int idx = blockIdx.x * blockDim.x + threadIdx.x;
    if (idx < n) st[idx] = 0u;
}

__global__ void __launch_bounds__(256) gf_prep_b(const int* __restrict__ P,
                                                 unsigned long long* __restrict__ bws) {
    const int idx = blockIdx.x * blockDim.x + threadIdx.x;
    unsigned p = (unsigned)P[idx] & 255u;
    unsigned rb[8] = {0,0,0,0,0,0,0,0};
#pragma unroll
    for (int t = 0; t < 8; ++t) {
#pragma unroll
        for (int r = 0; r < 8; ++r) rb[r] |= ((p >> r) & 1u) << t;
        p = xtime(p);
    }
    unsigned long long w = 0;
#pragma unroll
    for (int r = 0; r < 8; ++r) w |= (unsigned long long)rb[r] << (8 * r);
    bws[idx] = w;
}

__global__ void __launch_bounds__(256) gf_prep_a(const int* __restrict__ s,
                                                 unsigned* __restrict__ aws) {
    const int tid   = blockIdx.x * blockDim.x + threadIdx.x;
    const int chunk = tid >> 5, lane = tid & 31;
    const int half  = lane >> 4, m_row = lane & 15, j0 = chunk * 8;
    unsigned* outp = aws + (chunk * 32 + lane) * 16;
#pragma unroll
    for (int v = 0; v < 8; ++v) {
        const int jj = (v >> 2) * 4 + ((v >> 1) & 1) * 2 + half;
        const int tb = (v & 1) * 4;
        const unsigned sv0 = (unsigned)s[m_row * 256 + j0 + jj];
        const unsigned sv1 = (unsigned)s[(m_row + 16) * 256 + j0 + jj];
        outp[v]     = spread4((sv0 >> tb) & 0xFu);
        outp[8 + v] = spread4((sv1 >> tb) & 0xFu);
    }
}

template <bool PREP>
__global__ void __launch_bounds__(256) gf_gemm(const int* __restrict__ s,
                                               const int* __restrict__ P,
                                               const unsigned char* __restrict__ bws8,
                                               const unsigned* __restrict__ aws,
                                               unsigned* __restrict__ st) {
    const int lane  = threadIdx.x & 31;
    const int wave  = threadIdx.x >> 5;
    const int tile  = blockIdx.x * 8 + wave;
    const int n     = lane & 15;
    const int half  = lane >> 4;
    const int c     = tile * 16 + n;
    const int r     = c & 7;
    const int ki    = c >> 3;
    const int m_row = lane & 15;
    const int jjb   = half * 2;
    const int jjs[4] = { jjb / 2, jjb / 2 + 2, jjb / 2 + 4, jjb / 2 + 6 };

    v8i acc0 = {0,0,0,0,0,0,0,0};
    v8i acc1 = {0,0,0,0,0,0,0,0};

    for (int j0 = 0; j0 < 256; j0 += 8) {
        unsigned rb[4];
        if (PREP) {
#pragma unroll
            for (int u = 0; u < 4; ++u)
                rb[u] = bws8[(ki * 256 + j0 + jjs[u]) * 8 + r];
        } else {
#pragma unroll
            for (int u = 0; u < 4; ++u) {
                unsigned p = (unsigned)P[ki * 256 + j0 + jjs[u]] & 255u;
                unsigned a = 0;
#pragma unroll
                for (int t = 0; t < 8; ++t) { a |= ((p >> r) & 1u) << t; p = xtime(p); }
                rb[u] = a;
            }
        }
        v8i bfrag;
#pragma unroll
        for (int v = 0; v < 8; ++v) {
            const int u  = (v >> 2) * 2 + ((v >> 1) & 1);
            const int tb = (v & 1) * 4;
            bfrag[v] = (int)spread4((rb[u] >> tb) & 0xFu);
        }
        v8i afrag0, afrag1;
        if (PREP) {
            const unsigned* ap = aws + ((j0 >> 3) * 32 + lane) * 16;
            afrag0 = *(const v8i*)(ap);
            afrag1 = *(const v8i*)(ap + 8);
        } else {
#pragma unroll
            for (int v = 0; v < 8; ++v) {
                const int jj = (v >> 2) * 4 + ((v >> 1) & 1) * 2 + half;
                const int tb = (v & 1) * 4;
                const unsigned sv0 = (unsigned)s[m_row * 256 + j0 + jj];
                const unsigned sv1 = (unsigned)s[(m_row + 16) * 256 + j0 + jj];
                afrag0[v] = (int)spread4((sv0 >> tb) & 0xFu);
                afrag1[v] = (int)spread4((sv1 >> tb) & 0xFu);
            }
        }
        acc0 = __builtin_amdgcn_wmma_i32_16x16x64_iu8(false, afrag0, false, bfrag,
                                                      acc0, false, false);
        acc1 = __builtin_amdgcn_wmma_i32_16x16x64_iu8(false, afrag1, false, bfrag,
                                                      acc1, false, false);
    }

#pragma unroll
    for (int p = 0; p < 8; ++p) {
        const int b0 = p + half * 8;
        const int b1 = b0 + 16;
        atomicXor(&st[b0 * NKI + ki], ((unsigned)acc0[p] & 1u) << r);
        atomicXor(&st[b1 * NKI + ki], ((unsigned)acc1[p] & 1u) << r);
    }
}

__global__ void __launch_bounds__(256) gf_final(const int* __restrict__ m,
                                                const int* __restrict__ s,
                                                const unsigned* __restrict__ st,
                                                const int* __restrict__ table,
                                                float* __restrict__ out) {
    __shared__ unsigned sh[256];
    const int b = blockIdx.x;
    const int i = threadIdx.x;
    const unsigned s_i = (unsigned)s[b * 256 + i] & 255u;
    int total = 0;
    for (int k = 0; k < 16; ++k) {
        const unsigned v = st[b * NKI + k * 256 + i] & 255u;
        sh[i] = (unsigned)table[v * 256 + s_i];
        __syncthreads();
        for (int off = 128; off > 0; off >>= 1) {
            if (i < off) sh[i] ^= sh[i + off];
            __syncthreads();
        }
        if (i == 0) total += (int)sh[0] - m[b * 16 + k];
        __syncthreads();
    }
    if (i == 0) {
        const int res = 1 - total;
        out[b] = (float)(res > 0 ? res : 0);
    }
}

extern "C" void kernel_launch(void* const* d_in, const int* in_sizes, int n_in,
                              void* d_out, int out_size, void* d_ws, size_t ws_size,
                              hipStream_t stream) {
    const int* m     = (const int*)d_in[0];
    const int* s     = (const int*)d_in[1];
    const int* P     = (const int*)d_in[2];
    const int* table = (const int*)d_in[3];
    float* out = (float*)d_out;

    char* ws = (char*)d_ws;
    unsigned*           st  = (unsigned*)ws;
    unsigned long long* bws = (unsigned long long*)(ws + (512u << 10));
    unsigned*           aws = (unsigned*)(ws + (512u << 10) + (8u << 20));
    const size_t needed = (512u << 10) + (8u << 20) + (64u << 10);

    gf_zero<<<(32 * NKI + 255) / 256, 256, 0, stream>>>(st, 32 * NKI);
    if (ws_size >= needed) {
        gf_prep_b<<<(NKI * 256) / 256, 256, 0, stream>>>(P, bws);
        gf_prep_a<<<4, 256, 0, stream>>>(s, aws);
        gf_gemm<true><<<NTILE / 8, 256, 0, stream>>>(s, P, (const unsigned char*)bws,
                                                     aws, st);
    } else {
        gf_gemm<false><<<NTILE / 8, 256, 0, stream>>>(s, P, nullptr, nullptr, st);
    }
    gf_final<<<32, 256, 0, stream>>>(m, s, st, table, out);
}
